// NeuralFactorizationMachine_4071628997276
// MI455X (gfx1250) — matrix operand order repaired and checked by reading, not compiled, not run on hardware
//
#include <hip/hip_runtime.h>
#include <hip/hip_bf16.h>

#define B_DIM 1024
#define N_DIM 16384
#define D_DIM 512
#define O_DIM 4096

typedef __attribute__((ext_vector_type(16))) __bf16 v16bf;
typedef __attribute__((ext_vector_type(8)))  __bf16 v8bf;
typedef __attribute__((ext_vector_type(8)))  float  v8f;
typedef __attribute__((ext_vector_type(4)))  unsigned int u32x4;
typedef __attribute__((ext_vector_type(8)))  int i32x8;
typedef __attribute__((ext_vector_type(4)))  int i32x4;

#define WMMA_BF16(A, Bm, C) \
  __builtin_amdgcn_wmma_f32_16x16x32_bf16(false, (A), false, (Bm), (short)0, (C), false, false)

// ===========================================================================
// Fragment loaders (wave32 WMMA; this file's contiguous-K order, same for A and B)
//  A 16x32 bf16: lane l = row (l&15); 16 contiguous K at (l>>4)*16
//                -> vector elems 0..15.
//  B 32x16 bf16: lane l = column (l&15); 16 contiguous K at (l>>4)*16.
// ===========================================================================

// --- bf16-source loaders (fast path: pure b128 loads, no conversion) -------
__device__ __forceinline__ v16bf ld_frag_a_bf(const __bf16* __restrict__ p) {
  v8bf lo = *(const v8bf*)(p);
  v8bf hi = *(const v8bf*)(p + 8);
  v16bf f;
#pragma unroll
  for (int i = 0; i < 8; ++i) { f[i] = lo[i]; f[8 + i] = hi[i]; }
  return f;
}
__device__ __forceinline__ v16bf ld_frag_b_bf(const __bf16* __restrict__ p) {
  return *(const v16bf*)p;   // 32 contiguous bytes
}

// --- fp32-source loaders (fallback path: convert inline) -------------------
__device__ __forceinline__ v16bf ld_frag_a_f32(const float* __restrict__ p) {
  v16bf f;
#pragma unroll
  for (int i = 0; i < 8; ++i) f[i] = (__bf16)p[i];
#pragma unroll
  for (int i = 0; i < 8; ++i) f[8 + i] = (__bf16)p[8 + i];
  return f;
}
__device__ __forceinline__ void ld_frag_a_f32_dual(const float* __restrict__ p,
                                                   v16bf& f, v16bf& fq) {
  float t[16];
#pragma unroll
  for (int i = 0; i < 8; ++i) t[i] = p[i];
#pragma unroll
  for (int i = 0; i < 8; ++i) t[8 + i] = p[8 + i];
#pragma unroll
  for (int i = 0; i < 16; ++i) { f[i] = (__bf16)t[i]; fq[i] = (__bf16)(t[i] * t[i]); }
}
__device__ __forceinline__ v16bf ld_frag_b_f32(const float* __restrict__ p) {
  v16bf f;
#pragma unroll
  for (int i = 0; i < 16; ++i) f[i] = (__bf16)p[i];
  return f;
}
__device__ __forceinline__ void ld_frag_b_nn_f32_dual(const float* __restrict__ p,
                                                      int ld, v16bf& f, v16bf& fq) {
#pragma unroll
  for (int i = 0; i < 16; ++i) {
    float t = p[(size_t)i * ld];
    f[i] = (__bf16)t; fq[i] = (__bf16)(t * t);
  }
}

// ===========================================================================
// TDM: stage a (tile_dim1=128 rows x tile_dim0=32 K) bf16 tile of linear_w
// into LDS. D# built per CDNA5 ISA (2D tensor,
// groups 2/3 zero). amdgpu-toolchain 6-arg builtin form:
//   (u32x4 g0, i32x8 g1, i32x4 g2, i32x4 g3, i32x8 extra, i32 cpol)
// ===========================================================================
__device__ __forceinline__ void tdm_load_lw_tile(const __bf16* __restrict__ lw_bf,
                                                 int row0, int k0, unsigned lds_off) {
  unsigned long long ga = (unsigned long long)(size_t)lw_bf
                        + 2ull * ((unsigned long long)row0 * N_DIM + (unsigned)k0);
  u32x4 g0;
  g0[0] = 1u;                                            // count=1, user D#
  g0[1] = lds_off;                                       // lds_addr (bytes)
  g0[2] = (unsigned)(ga & 0xffffffffu);                  // global_addr[31:0]
  g0[3] = (unsigned)((ga >> 32) & 0x01ffffffu) | (2u << 30); // addr[56:32] | type=2
  i32x8 g1;
  g1[0] = (int)(1u << 16);                               // data_size=1 (2 bytes)
  g1[1] = (int)(((unsigned)N_DIM & 0xffffu) << 16);      // tensor_dim0[15:0]
  g1[2] = (int)((((unsigned)N_DIM >> 16) & 0xffffu)      // tensor_dim0[31:16]
              | (((unsigned)O_DIM & 0xffffu) << 16));    // tensor_dim1[15:0]
  g1[3] = (int)((((unsigned)O_DIM >> 16) & 0xffffu)      // tensor_dim1[31:16]
              | (32u << 16));                            // tile_dim0 = 32
  g1[4] = 128;                                           // tile_dim1=128, tile_dim2=0
  g1[5] = N_DIM;                                         // tensor_dim0_stride[31:0]
  g1[6] = 0;                                             // stride[47:32], dim1_stride lo
  g1[7] = 0;
  i32x4 gz4 = {0, 0, 0, 0};
  i32x8 gz8 = {0, 0, 0, 0, 0, 0, 0, 0};
  __builtin_amdgcn_tensor_load_to_lds(g0, g1, gz4, gz4, gz8, 0);
}

// ===========================================================================
// Pre-conversion kernels
// ===========================================================================
__global__ __launch_bounds__(256)
void cvt_bf16_kernel(const float* __restrict__ in, __bf16* __restrict__ out, int n) {
  int i = (blockIdx.x * 256 + threadIdx.x) * 4;
  if (i + 3 >= n) return;
  float4 v = *(const float4*)(in + i);
  out[i + 0] = (__bf16)v.x; out[i + 1] = (__bf16)v.y;
  out[i + 2] = (__bf16)v.z; out[i + 3] = (__bf16)v.w;
}

__global__ __launch_bounds__(256)
void cvt_bf16_pair_kernel(const float* __restrict__ in, __bf16* __restrict__ out,
                          __bf16* __restrict__ out_sq, int n) {
  int i = (blockIdx.x * 256 + threadIdx.x) * 4;
  if (i + 3 >= n) return;
  float4 v = *(const float4*)(in + i);
  out[i + 0] = (__bf16)v.x; out[i + 1] = (__bf16)v.y;
  out[i + 2] = (__bf16)v.z; out[i + 3] = (__bf16)v.w;
  out_sq[i + 0] = (__bf16)(v.x * v.x); out_sq[i + 1] = (__bf16)(v.y * v.y);
  out_sq[i + 2] = (__bf16)(v.z * v.z); out_sq[i + 3] = (__bf16)(v.w * v.w);
}

// emb (N,D) f32 -> embT, embT2 (D,N) bf16 via LDS-tiled transpose
__global__ __launch_bounds__(256)
void transpose_emb_kernel(const float* __restrict__ emb,
                          __bf16* __restrict__ embT, __bf16* __restrict__ embT2) {
  __shared__ float tile[32][33];
  const int n0 = blockIdx.x * 32, d0 = blockIdx.y * 32;
  const int tx = threadIdx.x & 31, ty = threadIdx.x >> 5;   // 32 x 8
#pragma unroll
  for (int r = 0; r < 32; r += 8)
    tile[ty + r][tx] = emb[(size_t)(n0 + ty + r) * D_DIM + d0 + tx];
  __syncthreads();
#pragma unroll
  for (int r = 0; r < 32; r += 8) {
    float v = tile[tx][ty + r];
    size_t idx = (size_t)(d0 + ty + r) * N_DIM + n0 + tx;
    embT[idx]  = (__bf16)v;
    embT2[idx] = (__bf16)(v * v);
  }
}

// ===========================================================================
// FAST PATH kernels (all operands pre-converted bf16, all GEMMs NT)
// ===========================================================================

// Kernel 1: accS = sae@emb, accQ = sae^2@emb^2 in one K pass; inter=0.5(s^2-q)
__global__ __launch_bounds__(128)
void fm_interaction_bf(const __bf16* __restrict__ sae_bf,
                       const __bf16* __restrict__ sae2_bf,
                       const __bf16* __restrict__ embT,
                       const __bf16* __restrict__ embT2,
                       __bf16* __restrict__ inter_bf) {
  const int lane = threadIdx.x & 31;
  const int wave = threadIdx.x >> 5;
  const int m0 = blockIdx.y * 64 + (wave >> 1) * 32;
  const int n0 = blockIdx.x * 64 + (wave & 1) * 32;
  const int ka = (lane >> 4) * 16, kb = (lane >> 4) * 16;
  const int rA0 = m0 + (lane & 15), rA1 = rA0 + 16;
  const int cB0 = n0 + (lane & 15), cB1 = cB0 + 16;

  v8f accS[2][2] = {}; v8f accQ[2][2] = {};
  for (int k0 = 0; k0 < N_DIM; k0 += 32) {
    v16bf a[2], aq[2], b[2], bq[2];
    a[0]  = ld_frag_a_bf(sae_bf  + (size_t)rA0 * N_DIM + k0 + ka);
    a[1]  = ld_frag_a_bf(sae_bf  + (size_t)rA1 * N_DIM + k0 + ka);
    aq[0] = ld_frag_a_bf(sae2_bf + (size_t)rA0 * N_DIM + k0 + ka);
    aq[1] = ld_frag_a_bf(sae2_bf + (size_t)rA1 * N_DIM + k0 + ka);
    b[0]  = ld_frag_b_bf(embT  + (size_t)cB0 * N_DIM + k0 + kb);
    b[1]  = ld_frag_b_bf(embT  + (size_t)cB1 * N_DIM + k0 + kb);
    bq[0] = ld_frag_b_bf(embT2 + (size_t)cB0 * N_DIM + k0 + kb);
    bq[1] = ld_frag_b_bf(embT2 + (size_t)cB1 * N_DIM + k0 + kb);
#pragma unroll
    for (int i = 0; i < 2; ++i)
#pragma unroll
      for (int j = 0; j < 2; ++j) {
        accS[i][j] = WMMA_BF16(a[i],  b[j],  accS[i][j]);
        accQ[i][j] = WMMA_BF16(aq[i], bq[j], accQ[i][j]);
      }
  }
  const int rbase = (lane >> 4) * 8, cl = lane & 15;
#pragma unroll
  for (int i = 0; i < 2; ++i)
#pragma unroll
    for (int j = 0; j < 2; ++j)
#pragma unroll
      for (int e = 0; e < 8; ++e) {
        float s = accS[i][j][e], q = accQ[i][j][e];
        inter_bf[(size_t)(m0 + 16 * i + rbase + e) * D_DIM + (n0 + 16 * j + cl)] =
            (__bf16)(0.5f * (s * s - q));
      }
}

// Kernel 2: h = relu(inter @ w1^T + b1), K = D
__global__ __launch_bounds__(128)
void mlp1_bf(const __bf16* __restrict__ inter_bf, const __bf16* __restrict__ w1_bf,
             const float* __restrict__ b1, __bf16* __restrict__ h_bf) {
  const int lane = threadIdx.x & 31;
  const int wave = threadIdx.x >> 5;
  const int m0 = blockIdx.y * 64 + (wave >> 1) * 32;
  const int n0 = blockIdx.x * 64 + (wave & 1) * 32;
  const int ka = (lane >> 4) * 16, kb = (lane >> 4) * 16;
  const int rA0 = m0 + (lane & 15), rA1 = rA0 + 16;
  const int cB0 = n0 + (lane & 15), cB1 = cB0 + 16;

  v8f acc[2][2] = {};
  for (int k0 = 0; k0 < D_DIM; k0 += 32) {
    v16bf a[2], b[2];
    a[0] = ld_frag_a_bf(inter_bf + (size_t)rA0 * D_DIM + k0 + ka);
    a[1] = ld_frag_a_bf(inter_bf + (size_t)rA1 * D_DIM + k0 + ka);
    b[0] = ld_frag_b_bf(w1_bf + (size_t)cB0 * D_DIM + k0 + kb);
    b[1] = ld_frag_b_bf(w1_bf + (size_t)cB1 * D_DIM + k0 + kb);
#pragma unroll
    for (int i = 0; i < 2; ++i)
#pragma unroll
      for (int j = 0; j < 2; ++j)
        acc[i][j] = WMMA_BF16(a[i], b[j], acc[i][j]);
  }
  const int rbase = (lane >> 4) * 8, cl = lane & 15;
#pragma unroll
  for (int i = 0; i < 2; ++i)
#pragma unroll
    for (int j = 0; j < 2; ++j) {
      const int col = n0 + 16 * j + cl;
      const float bias = b1[col];
#pragma unroll
      for (int e = 0; e < 8; ++e) {
        float v = acc[i][j][e] + bias;
        h_bf[(size_t)(m0 + 16 * i + rbase + e) * D_DIM + col] = (__bf16)(v > 0.f ? v : 0.f);
      }
    }
}

// Kernel 3: out = sae@lw^T + h@w2^T + lb + b2.
// Dominant K-phase streams lw_bf tiles through LDS via TDM, double-buffered.
// Block = 64(M) x 128(N); 4 waves, each 32x64 (2x4 fragments).
#define LWTILE_ELEMS (128 * 32)
__global__ __launch_bounds__(128)
void output_bf(const __bf16* __restrict__ sae_bf, const __bf16* __restrict__ lw_bf,
               const float* __restrict__ lb, const __bf16* __restrict__ h_bf,
               const __bf16* __restrict__ w2_bf, const float* __restrict__ b2,
               float* __restrict__ out) {
  __shared__ __bf16 ldsB[2][LWTILE_ELEMS];
  const int lane = threadIdx.x & 31;
  const int wave = threadIdx.x >> 5;
  const int m0 = blockIdx.y * 64 + (wave >> 1) * 32;   // wave M base (global)
  const int nblk = blockIdx.x * 128;                   // block N base (global)
  const int nw = (wave & 1) * 64;                      // wave N base (block-local)
  const int ka = (lane >> 4) * 16, kb = (lane >> 4) * 16;
  const int rA0 = m0 + (lane & 15), rA1 = rA0 + 16;

  v8f acc[2][4] = {};

  // ---- Phase 1: sae @ linear_w^T, K = N_DIM, B tiles via TDM ----
  const unsigned ldsOff[2] = { (unsigned)(size_t)&ldsB[0][0],
                               (unsigned)(size_t)&ldsB[1][0] };
  if (wave == 0) {
    tdm_load_lw_tile(lw_bf, nblk, 0, ldsOff[0]);
    __builtin_amdgcn_s_wait_tensorcnt(0);
  }
  __syncthreads();

  int buf = 0;
  for (int k0 = 0; k0 < N_DIM; k0 += 32) {
    const bool more = (k0 + 32) < N_DIM;
    if (wave == 0 && more)
      tdm_load_lw_tile(lw_bf, nblk, k0 + 32, ldsOff[buf ^ 1]);

    v16bf a[2], b[4];
    a[0] = ld_frag_a_bf(sae_bf + (size_t)rA0 * N_DIM + k0 + ka);
    a[1] = ld_frag_a_bf(sae_bf + (size_t)rA1 * N_DIM + k0 + ka);
#pragma unroll
    for (int j = 0; j < 4; ++j) {
      const int colLocal = nw + 16 * j + (lane & 15);  // lw row, block-local
      b[j] = ld_frag_b_bf(&ldsB[buf][colLocal * 32 + kb]);
    }
#pragma unroll
    for (int i = 0; i < 2; ++i)
#pragma unroll
      for (int j = 0; j < 4; ++j)
        acc[i][j] = WMMA_BF16(a[i], b[j], acc[i][j]);

    if (wave == 0 && more)
      __builtin_amdgcn_s_wait_tensorcnt(0);
    __syncthreads();
    buf ^= 1;
  }

  // ---- Phase 2: h @ mlp2_w^T, K = D_DIM, direct global ----
  for (int k0 = 0; k0 < D_DIM; k0 += 32) {
    v16bf a[2], b[4];
    a[0] = ld_frag_a_bf(h_bf + (size_t)rA0 * D_DIM + k0 + ka);
    a[1] = ld_frag_a_bf(h_bf + (size_t)rA1 * D_DIM + k0 + ka);
#pragma unroll
    for (int j = 0; j < 4; ++j) {
      const int col = nblk + nw + 16 * j + (lane & 15);
      b[j] = ld_frag_b_bf(w2_bf + (size_t)col * D_DIM + k0 + kb);
    }
#pragma unroll
    for (int i = 0; i < 2; ++i)
#pragma unroll
      for (int j = 0; j < 4; ++j)
        acc[i][j] = WMMA_BF16(a[i], b[j], acc[i][j]);
  }

  const int rbase = (lane >> 4) * 8, cl = lane & 15;
#pragma unroll
  for (int i = 0; i < 2; ++i)
#pragma unroll
    for (int j = 0; j < 4; ++j) {
      const int col = nblk + nw + 16 * j + cl;
      const float bias = lb[col] + b2[col];
#pragma unroll
      for (int e = 0; e < 8; ++e)
        out[(size_t)(m0 + 16 * i + rbase + e) * O_DIM + col] = acc[i][j][e] + bias;
    }
}

// ===========================================================================
// FALLBACK PATH (round-1 style, fp32 sources, inline convert; used when
// ws_size is too small for the bf16 staging buffers)
// ===========================================================================
__global__ __launch_bounds__(128)
void fm_interaction_f32(const float* __restrict__ sae, const float* __restrict__ emb,
                        float* __restrict__ inter) {
  const int lane = threadIdx.x & 31;
  const int wave = threadIdx.x >> 5;
  const int m0 = blockIdx.y * 64 + (wave >> 1) * 32;
  const int n0 = blockIdx.x * 64 + (wave & 1) * 32;
  const int ka = (lane >> 4) * 16, kb = (lane >> 4) * 16;
  const int rA0 = m0 + (lane & 15), rA1 = rA0 + 16;
  const int cB0 = n0 + (lane & 15), cB1 = cB0 + 16;

  v8f accS[2][2] = {}; v8f accQ[2][2] = {};
  for (int k0 = 0; k0 < N_DIM; k0 += 32) {
    v16bf a[2], aq[2], b[2], bq[2];
    ld_frag_a_f32_dual(sae + (size_t)rA0 * N_DIM + k0 + ka, a[0], aq[0]);
    ld_frag_a_f32_dual(sae + (size_t)rA1 * N_DIM + k0 + ka, a[1], aq[1]);
    ld_frag_b_nn_f32_dual(emb + (size_t)(k0 + kb) * D_DIM + cB0, D_DIM, b[0], bq[0]);
    ld_frag_b_nn_f32_dual(emb + (size_t)(k0 + kb) * D_DIM + cB1, D_DIM, b[1], bq[1]);
#pragma unroll
    for (int i = 0; i < 2; ++i)
#pragma unroll
      for (int j = 0; j < 2; ++j) {
        accS[i][j] = WMMA_BF16(a[i],  b[j],  accS[i][j]);
        accQ[i][j] = WMMA_BF16(aq[i], bq[j], accQ[i][j]);
      }
  }
  const int rbase = (lane >> 4) * 8, cl = lane & 15;
#pragma unroll
  for (int i = 0; i < 2; ++i)
#pragma unroll
    for (int j = 0; j < 2; ++j)
#pragma unroll
      for (int e = 0; e < 8; ++e) {
        float s = accS[i][j][e], q = accQ[i][j][e];
        inter[(size_t)(m0 + 16 * i + rbase + e) * D_DIM + (n0 + 16 * j + cl)] =
            0.5f * (s * s - q);
      }
}

__global__ __launch_bounds__(128)
void mlp1_f32(const float* __restrict__ inter, const float* __restrict__ w1,
              const float* __restrict__ b1, float* __restrict__ h) {
  const int lane = threadIdx.x & 31;
  const int wave = threadIdx.x >> 5;
  const int m0 = blockIdx.y * 64 + (wave >> 1) * 32;
  const int n0 = blockIdx.x * 64 + (wave & 1) * 32;
  const int ka = (lane >> 4) * 16, kb = (lane >> 4) * 16;
  const int rA0 = m0 + (lane & 15), rA1 = rA0 + 16;
  const int cB0 = n0 + (lane & 15), cB1 = cB0 + 16;

  v8f acc[2][2] = {};
  for (int k0 = 0; k0 < D_DIM; k0 += 32) {
    v16bf a[2], b[2];
    a[0] = ld_frag_a_f32(inter + (size_t)rA0 * D_DIM + k0 + ka);
    a[1] = ld_frag_a_f32(inter + (size_t)rA1 * D_DIM + k0 + ka);
    b[0] = ld_frag_b_f32(w1 + (size_t)cB0 * D_DIM + k0 + kb);
    b[1] = ld_frag_b_f32(w1 + (size_t)cB1 * D_DIM + k0 + kb);
#pragma unroll
    for (int i = 0; i < 2; ++i)
#pragma unroll
      for (int j = 0; j < 2; ++j)
        acc[i][j] = WMMA_BF16(a[i], b[j], acc[i][j]);
  }
  const int rbase = (lane >> 4) * 8, cl = lane & 15;
#pragma unroll
  for (int i = 0; i < 2; ++i)
#pragma unroll
    for (int j = 0; j < 2; ++j) {
      const int col = n0 + 16 * j + cl;
      const float bias = b1[col];
#pragma unroll
      for (int e = 0; e < 8; ++e) {
        float v = acc[i][j][e] + bias;
        h[(size_t)(m0 + 16 * i + rbase + e) * D_DIM + col] = v > 0.0f ? v : 0.0f;
      }
    }
}

__global__ __launch_bounds__(128)
void output_f32(const float* __restrict__ sae, const float* __restrict__ lw,
                const float* __restrict__ lb, const float* __restrict__ h,
                const float* __restrict__ w2, const float* __restrict__ b2,
                float* __restrict__ out) {
  const int lane = threadIdx.x & 31;
  const int wave = threadIdx.x >> 5;
  const int m0 = blockIdx.y * 64 + (wave >> 1) * 32;
  const int n0 = blockIdx.x * 64 + (wave & 1) * 32;
  const int ka = (lane >> 4) * 16, kb = (lane >> 4) * 16;
  const int rA0 = m0 + (lane & 15), rA1 = rA0 + 16;
  const int cB0 = n0 + (lane & 15), cB1 = cB0 + 16;

  v8f acc[2][2] = {};
  for (int k0 = 0; k0 < N_DIM; k0 += 32) {
    v16bf a[2], b[2];
    a[0] = ld_frag_a_f32(sae + (size_t)rA0 * N_DIM + k0 + ka);
    a[1] = ld_frag_a_f32(sae + (size_t)rA1 * N_DIM + k0 + ka);
    b[0] = ld_frag_b_f32(lw + (size_t)cB0 * N_DIM + k0 + kb);
    b[1] = ld_frag_b_f32(lw + (size_t)cB1 * N_DIM + k0 + kb);
#pragma unroll
    for (int i = 0; i < 2; ++i)
#pragma unroll
      for (int j = 0; j < 2; ++j)
        acc[i][j] = WMMA_BF16(a[i], b[j], acc[i][j]);
  }
  for (int k0 = 0; k0 < D_DIM; k0 += 32) {
    v16bf a[2], b[2];
    a[0] = ld_frag_a_f32(h + (size_t)rA0 * D_DIM + k0 + ka);
    a[1] = ld_frag_a_f32(h + (size_t)rA1 * D_DIM + k0 + ka);
    b[0] = ld_frag_b_f32(w2 + (size_t)cB0 * D_DIM + k0 + kb);
    b[1] = ld_frag_b_f32(w2 + (size_t)cB1 * D_DIM + k0 + kb);
#pragma unroll
    for (int i = 0; i < 2; ++i)
#pragma unroll
      for (int j = 0; j < 2; ++j)
        acc[i][j] = WMMA_BF16(a[i], b[j], acc[i][j]);
  }
  const int rbase = (lane >> 4) * 8, cl = lane & 15;
#pragma unroll
  for (int i = 0; i < 2; ++i)
#pragma unroll
    for (int j = 0; j < 2; ++j) {
      const int col = n0 + 16 * j + cl;
      const float bias = lb[col] + b2[col];
#pragma unroll
      for (int e = 0; e < 8; ++e)
        out[(size_t)(m0 + 16 * i + rbase + e) * O_DIM + col] = acc[i][j][e] + bias;
    }
}

// ===========================================================================
extern "C" void kernel_launch(void* const* d_in, const int* in_sizes, int n_in,
                              void* d_out, int out_size, void* d_ws, size_t ws_size,
                              hipStream_t stream) {
  const float* sae = (const float*)d_in[0];   // (B, N)
  const float* emb = (const float*)d_in[1];   // (N, D)
  const float* lw  = (const float*)d_in[2];   // (O, N)
  const float* lb  = (const float*)d_in[3];   // (O,)
  const float* w1  = (const float*)d_in[4];   // (D, D)
  const float* b1  = (const float*)d_in[5];   // (D,)
  const float* w2  = (const float*)d_in[6];   // (O, D)
  const float* b2  = (const float*)d_in[7];   // (O,)
  float* out = (float*)d_out;                 // (B, O)

  const size_t nBN = (size_t)B_DIM * N_DIM;   // 16.8M
  const size_t nON = (size_t)O_DIM * N_DIM;   // 67.1M
  const size_t nDN = (size_t)D_DIM * N_DIM;   // 8.4M
  const size_t nDD = (size_t)D_DIM * D_DIM;
  const size_t nOD = (size_t)O_DIM * D_DIM;
  const size_t nBD = (size_t)B_DIM * D_DIM;

  const size_t needBytes =
      2 * (nBN * 2 + nON + nDN * 2 + nDD + nOD + nBD * 2);

  if (ws_size >= needBytes) {
    // -------- fast path: pre-convert everything to bf16 once --------
    __bf16* p = (__bf16*)d_ws;
    __bf16* sae_bf  = p; p += nBN;
    __bf16* sae2_bf = p; p += nBN;
    __bf16* lw_bf   = p; p += nON;
    __bf16* embT    = p; p += nDN;
    __bf16* embT2   = p; p += nDN;
    __bf16* w1_bf   = p; p += nDD;
    __bf16* w2_bf   = p; p += nOD;
    __bf16* inter_bf= p; p += nBD;
    __bf16* h_bf    = p; p += nBD;

    cvt_bf16_pair_kernel<<<(unsigned)(nBN / 1024), 256, 0, stream>>>(sae, sae_bf, sae2_bf, (int)nBN);
    cvt_bf16_kernel<<<(unsigned)(nON / 1024), 256, 0, stream>>>(lw, lw_bf, (int)nON);
    cvt_bf16_kernel<<<(unsigned)(nDD / 1024), 256, 0, stream>>>(w1, w1_bf, (int)nDD);
    cvt_bf16_kernel<<<(unsigned)(nOD / 1024), 256, 0, stream>>>(w2, w2_bf, (int)nOD);
    transpose_emb_kernel<<<dim3(N_DIM / 32, D_DIM / 32), 256, 0, stream>>>(emb, embT, embT2);

    fm_interaction_bf<<<dim3(D_DIM / 64, B_DIM / 64), 128, 0, stream>>>(
        sae_bf, sae2_bf, embT, embT2, inter_bf);
    mlp1_bf<<<dim3(D_DIM / 64, B_DIM / 64), 128, 0, stream>>>(inter_bf, w1_bf, b1, h_bf);
    output_bf<<<dim3(O_DIM / 128, B_DIM / 64), 128, 0, stream>>>(
        sae_bf, lw_bf, lb, h_bf, w2_bf, b2, out);
  } else {
    // -------- fallback: inline-convert kernels (4 MB workspace) --------
    float* inter = (float*)d_ws;
    float* h = inter + nBD;
    fm_interaction_f32<<<dim3(D_DIM / 64, B_DIM / 64), 128, 0, stream>>>(sae, emb, inter);
    mlp1_f32<<<dim3(D_DIM / 64, B_DIM / 64), 128, 0, stream>>>(inter, w1, b1, h);
    output_f32<<<dim3(O_DIM / 64, B_DIM / 64), 128, 0, stream>>>(sae, lw, lb, h, w2, b2, out);
  }
}
